// SetTransformerHSP_sb_seed_85925115723966
// MI455X (gfx1250) — matrix operand order repaired and checked by reading, not compiled, not run on hardware
//
#include <hip/hip_runtime.h>

typedef __bf16 bf16_t;
typedef __attribute__((ext_vector_type(16))) __bf16 v16bf;
typedef __attribute__((ext_vector_type(8)))  float  v8f;
typedef __attribute__((ext_vector_type(4)))  unsigned int uint4v;
typedef __attribute__((ext_vector_type(4)))  unsigned int tdm_u4;
typedef __attribute__((ext_vector_type(8)))  int tdm_i8;
typedef __attribute__((ext_vector_type(4)))  int tdm_i4;

#define HAVE_TDM __has_builtin(__builtin_amdgcn_tensor_load_to_lds)

union V16 { uint4v q[2]; v16bf v; };

static __device__ inline uint4v uz4() { uint4v z = {0u,0u,0u,0u}; return z; }
static __device__ inline v8f vzero8() {
  v8f z = {0.f,0.f,0.f,0.f,0.f,0.f,0.f,0.f};
  return z;
}

static __device__ inline v8f wmma_bf16(v16bf a, v16bf b, v8f c) {
  return __builtin_amdgcn_wmma_f32_16x16x32_bf16(false, a, false, b,
                                                 (short)0, c, false, false);
}

// A fragment 16x32 (MxK), row-major, ld%8==0, 16B-aligned: two b128 loads.
static __device__ inline v16bf frag_a16x32(const bf16_t* __restrict__ p, int ld) {
  const int lane = threadIdx.x & 31;
  const bf16_t* row = p + (size_t)(lane & 15) * ld + ((lane >> 4) << 3);
  V16 u;
  u.q[0] = *(const uint4v*)(row);
  u.q[1] = *(const uint4v*)(row + 16);
  return u.v;
}

// B fragment 32x16 (KxN), "n-major" storage: element (k,n) = p[n*ld + k].
static __device__ inline v16bf frag_b16n(const bf16_t* __restrict__ p, int ld) {
  const int lane = threadIdx.x & 31;
  const bf16_t* row = p + (size_t)(lane & 15) * ld + ((lane >> 4) << 3);
  V16 u;
  u.q[0] = *(const uint4v*)(row);
  u.q[1] = *(const uint4v*)(row + 16);
  return u.v;
}

// ---------------------------------------------------------------- TDM issue
// 2D tensor->LDS DMA per ISA D# (group0 + group1); dsz = data_size code
// (0=1B, 1=2B); optional LDS row padding; OOB rows/cols zero-filled.
#if HAVE_TDM
static __device__ __forceinline__ void tdm_load_2d(
    unsigned lds_addr, const void* gaddr,
    unsigned dim0, unsigned dim1, unsigned stride0,
    unsigned tile0, unsigned tile1,
    unsigned dsz, unsigned padi, unsigned pada, unsigned pen) {
  const unsigned long long ga = (unsigned long long)(uintptr_t)gaddr;
  tdm_u4 g0 = { 1u,                                   // count=1, user mode
                lds_addr,
                (unsigned)(ga & 0xFFFFFFFFu),
                (unsigned)((ga >> 32) & 0x01FFFFFFu) | (2u << 30) };  // type=2
  const unsigned w0 = (dsz << 16) | (pen << 20) | (padi << 22) | (pada << 25);
  tdm_i8 g1 = { (int)w0,
                (int)((dim0 & 0xFFFFu) << 16),
                (int)(((dim0 >> 16) & 0xFFFFu) | ((dim1 & 0xFFFFu) << 16)),
                (int)(((dim1 >> 16) & 0xFFFFu) | ((tile0 & 0xFFFFu) << 16)),
                (int)(tile1 & 0xFFFFu),
                (int)stride0, 0, 0 };
  tdm_i4 z4 = {0,0,0,0};
#if __clang_major__ >= 23
  tdm_i8 z8 = {0,0,0,0,0,0,0,0};
  __builtin_amdgcn_tensor_load_to_lds(g0, g1, z4, z4, z8, 0);
#else
  __builtin_amdgcn_tensor_load_to_lds(g0, g1, z4, z4, 0);
#endif
}
#endif

// ---------------------------------------------------------------- small kernels
__global__ void k_cvt(const float* __restrict__ s, bf16_t* __restrict__ d, int n) {
  int i = blockIdx.x * blockDim.x + threadIdx.x;
  if (i < n) d[i] = (bf16_t)s[i];
}

__global__ void k_cvtPad(const float* __restrict__ s, bf16_t* __restrict__ d,
                         int M, int K, int Kpad) {
  int i = blockIdx.x * blockDim.x + threadIdx.x;
  if (i >= M * Kpad) return;
  int r = i / Kpad, k = i - r * Kpad;
  d[i] = (k < K) ? (bf16_t)s[(size_t)r * K + k] : (bf16_t)0.0f;
}

__global__ void k_cvtT(const float* __restrict__ s, bf16_t* __restrict__ d,
                       int K, int N, int Kpad) {
  int i = blockIdx.x * blockDim.x + threadIdx.x;
  if (i >= N * Kpad) return;
  int n = i / Kpad, k = i - n * Kpad;
  d[i] = (k < K) ? (bf16_t)s[(size_t)k * N + n] : (bf16_t)0.0f;
}

__global__ void k_adj8(const float* __restrict__ s, unsigned char* __restrict__ d, int n) {
  int i = blockIdx.x * blockDim.x + threadIdx.x;
  if (i < n) d[i] = (unsigned char)(s[i] != 0.0f ? 1 : 0);
}

__global__ void k_zero16(bf16_t* __restrict__ d, int n) {
  int i = blockIdx.x * blockDim.x + threadIdx.x;
  if (i < n) d[i] = (bf16_t)0.0f;
}

// ---------------------------------------------------------------- GEMM
// C[M,N] = epi( A[M,K] @ W[K,N] + bias[N] ), K%32==0, A ld=K, Wt[N,K].
// EPI 0: bf16; 1: resid+relu bf16; 2: f32; 3: V^T key-permuted scatter.
// Tile 128x64, BK=32, 4 waves (32 rows each). TDM double-buffered staging.
#define GLDA 40
#define WLD  40
template <int EPI>
__launch_bounds__(128)
__global__ void k_gemm(const bf16_t* __restrict__ A, const bf16_t* __restrict__ Wt,
                       const float* __restrict__ bias,
                       const bf16_t* __restrict__ resid,
                       bf16_t* __restrict__ Cbf, float* __restrict__ Cf,
                       bf16_t* __restrict__ CvT, int kb_shift, int mpad,
                       int M, int N, int K)
{
  __shared__ __align__(16) bf16_t As[2][128 * GLDA];
  __shared__ __align__(16) bf16_t Ws[2][64 * WLD];
  const int tid  = threadIdx.x;
  const int lane = tid & 31;
  const int wave = tid >> 5;
  const int m0 = blockIdx.x * 128;
  const int n0 = blockIdx.y * 64;
  const int nk = K >> 5;

  v8f acc[2][4];
#pragma unroll
  for (int t = 0; t < 2; ++t)
#pragma unroll
    for (int j = 0; j < 4; ++j) acc[t][j] = vzero8();

#if HAVE_TDM
  auto issue = [&](int kt, int buf) {
    const int k0 = kt * 32;
    tdm_load_2d((unsigned)(uintptr_t)&As[buf][0], &A[(size_t)m0 * K + k0],
                (unsigned)(K - k0), (unsigned)(M - m0), (unsigned)K,
                32u, 128u, 1u, 3u, 3u, 1u);      // pad 4 DW after each 16 DW row
    tdm_load_2d((unsigned)(uintptr_t)&Ws[buf][0], &Wt[(size_t)n0 * K + k0],
                (unsigned)(K - k0), (unsigned)(N - n0), (unsigned)K,
                32u, 64u, 1u, 3u, 3u, 1u);
  };
  if (wave == 0) issue(0, 0);
#else
  auto stage = [&](int kt, int buf) {
    const int k0 = kt * 32;
    for (int i = tid; i < 512; i += 128) {       // A 128x32
      const int r = i >> 2, c8 = (i & 3) << 3;
      const int rc = (m0 + r < M) ? (m0 + r) : (M - 1);
      *(uint4v*)&As[buf][r * GLDA + c8] = *(const uint4v*)&A[(size_t)rc * K + k0 + c8];
    }
    for (int i = tid; i < 256; i += 128) {       // Wt 64x32
      const int c = i >> 2, k8 = (i & 3) << 3;
      *(uint4v*)&Ws[buf][c * WLD + k8] = *(const uint4v*)&Wt[(size_t)(n0 + c) * K + k0 + k8];
    }
  };
  stage(0, 0);
#endif

  for (int kt = 0; kt < nk; ++kt) {
    const int cur = kt & 1;
#if HAVE_TDM
    if (wave == 0) {
      if (kt + 1 < nk) {
        issue(kt + 1, cur ^ 1);
        __builtin_amdgcn_s_wait_tensorcnt(2);
      } else {
        __builtin_amdgcn_s_wait_tensorcnt(0);
      }
    }
#else
    if (kt + 1 < nk) stage(kt + 1, cur ^ 1);
#endif
    __syncthreads();

    const v16bf af0 = frag_a16x32(&As[cur][(wave * 32) * GLDA], GLDA);
    const v16bf af1 = frag_a16x32(&As[cur][(wave * 32 + 16) * GLDA], GLDA);
#pragma unroll
    for (int j = 0; j < 4; ++j) {
      const v16bf bfr = frag_b16n(&Ws[cur][(j * 16) * WLD], WLD);
      acc[0][j] = wmma_bf16(af0, bfr, acc[0][j]);
      acc[1][j] = wmma_bf16(af1, bfr, acc[1][j]);
    }
    __syncthreads();
  }

  const int half = lane >> 4, nn = lane & 15;
#pragma unroll
  for (int t = 0; t < 2; ++t) {
#pragma unroll
    for (int j = 0; j < 4; ++j) {
#pragma unroll
      for (int r = 0; r < 8; ++r) {
        const int row = m0 + wave * 32 + t * 16 + r + 8 * half;
        if (row >= M) continue;
        const int col = n0 + j * 16 + nn;
        const float v = acc[t][j][r] + bias[col];
        if (EPI == 0) {
          Cbf[(size_t)row * N + col] = (bf16_t)v;
        } else if (EPI == 1) {
          Cbf[(size_t)row * N + col] =
              (bf16_t)((float)resid[(size_t)row * N + col] + fmaxf(v, 0.0f));
        } else if (EPI == 2) {
          Cf[(size_t)row * N + col] = v;
        } else {
          const int bidx = row >> kb_shift;
          const int g = row & ((1 << kb_shift) - 1);
          const int pos = (g & ~31) + (((g & 15) << 1) | ((g >> 4) & 1));
          CvT[((size_t)bidx * N + col) * mpad + pos] = (bf16_t)v;
        }
      }
    }
  }
}

// ---------------------------------------------------------------- attention
// S = (Q K^T)/sqrt(512) + alpha_h*(1-adj); P = softmax; O = Q + P V.
// V is transposed + key-permuted (slot 2t<->key t, 2t+1<->key t+16 per 32-key
// tile): P packs as b32 LDS stores, all B-fragments contiguous b128, row-sums
// via an extra WMMA against a ones B-matrix. K, V^T and the 64x32 adj u8 tile
// all stream through the Tensor Data Mover, double-buffered (3 D#s in flight).
#define ALD 136
#define VLD 40
__launch_bounds__(128)
__global__ void k_attn(const bf16_t* __restrict__ Qp, const bf16_t* __restrict__ Kp,
                       const bf16_t* __restrict__ VpT,
                       const unsigned char* __restrict__ adj,
                       bf16_t* __restrict__ O,
                       int n, int m, int mpad, int adjN, int row0, int col0)
{
  __shared__ __align__(16) bf16_t Ks[2][32 * ALD];
  __shared__ __align__(16) bf16_t VsT[2][128 * VLD];
  __shared__ __align__(16) unsigned char AdjS[2][64 * 32];
  __shared__ __align__(16) bf16_t Pst[4][16 * 32];

  const int tid  = threadIdx.x;
  const int lane = tid & 31;
  const int wave = tid >> 5;
  const int b = blockIdx.y >> 2;
  const int h = blockIdx.y & 3;
  const float alpha = (h == 0) ? -0.0f : (h == 1) ? -1.0f : (h == 2) ? -5.0f : -1e9f;
  const float scale = 0.044194173824159216f;   // 1/sqrt(512)

  const int qblk   = blockIdx.x * 64;
  const int qbase  = qblk + wave * 16;
  const bool qvalid = qbase < n;
  const int qclamp = qvalid ? qbase : 0;

  const bf16_t* Qhead = Qp + (size_t)b * n * 512 + h * 128;
  const unsigned char* adjb = adj + (size_t)b * adjN * adjN;

  v16bf qf[4];
#pragma unroll
  for (int kk = 0; kk < 4; ++kk)
    qf[kk] = frag_a16x32(Qhead + (size_t)qclamp * 512 + kk * 32, 512);

  v16bf onesf;
#pragma unroll
  for (int i = 0; i < 16; ++i) onesf[i] = (bf16_t)1.0f;

  v8f oacc[8];
#pragma unroll
  for (int j = 0; j < 8; ++j) oacc[j] = vzero8();
  v8f lacc = vzero8();
  float mst[8];
#pragma unroll
  for (int r = 0; r < 8; ++r) mst[r] = -1e30f;

  const int half = lane >> 4, nn = lane & 15;
  bf16_t* myP = &Pst[wave][0];
  unsigned int* myP32 = (unsigned int*)myP;
  const int nt = (m + 31) >> 5;

#if HAVE_TDM
  auto issue = [&](int it, int buf) {
    const int mt = it * 32;
    tdm_load_2d((unsigned)(uintptr_t)&Ks[buf][0],
                &Kp[((size_t)b * m + mt) * 512 + h * 128],
                512u, (unsigned)(m - mt), 512u, 128u, 32u, 1u, 5u, 3u, 1u);
    tdm_load_2d((unsigned)(uintptr_t)&VsT[buf][0],
                &VpT[((size_t)b * 512 + h * 128) * (size_t)mpad + mt],
                (unsigned)(mpad - mt), 128u, (unsigned)mpad, 32u, 128u, 1u, 3u, 3u, 1u);
    tdm_load_2d((unsigned)(uintptr_t)&AdjS[buf][0],
                adjb + (size_t)(row0 + qblk) * adjN + col0 + mt,
                (unsigned)(adjN - (col0 + mt)), (unsigned)(adjN - (row0 + qblk)),
                (unsigned)adjN, 32u, 64u, 0u, 0u, 0u, 0u);
  };
  if (wave == 0) issue(0, 0);
#else
  auto stage = [&](int it, int buf) {
    const int mt = it * 32;
    for (int i = tid; i < 512; i += 128) {       // K 32x128 (clamped rows)
      const int r = i >> 4, c8 = (i & 15) << 3;
      const int key = (mt + r < m) ? (mt + r) : (m - 1);
      *(uint4v*)&Ks[buf][r * ALD + c8] =
          *(const uint4v*)&Kp[((size_t)b * m + key) * 512 + h * 128 + c8];
    }
    for (int i = tid; i < 512; i += 128) {       // V^T 128x32
      const int f = i >> 2, c8 = (i & 3) << 3;
      *(uint4v*)&VsT[buf][f * VLD + c8] =
          *(const uint4v*)&VpT[((size_t)b * 512 + h * 128 + f) * mpad + mt + c8];
    }
    for (int i = tid; i < 64 * 32; i += 128) {   // adj 64x32 u8
      const int r = i >> 5, c = i & 31;
      const int ar = row0 + qblk + r, ac = col0 + mt + c;
      AdjS[buf][i] = (ar < adjN && ac < adjN) ? adjb[(size_t)ar * adjN + ac]
                                              : (unsigned char)0;
    }
  };
  stage(0, 0);
#endif

  for (int it = 0; it < nt; ++it) {
    const int cur = it & 1;
    const int mt = it * 32;
#if HAVE_TDM
    if (wave == 0) {
      if (it + 1 < nt) {
        issue(it + 1, cur ^ 1);
        __builtin_amdgcn_s_wait_tensorcnt(3);
      } else {
        __builtin_amdgcn_s_wait_tensorcnt(0);
      }
    }
#else
    if (it + 1 < nt) stage(it + 1, cur ^ 1);
#endif
    __syncthreads();

    v8f s0 = vzero8(), s1 = vzero8();
#pragma unroll
    for (int kk = 0; kk < 4; ++kk) {
      s0 = wmma_bf16(qf[kk], frag_b16n(&Ks[cur][kk * 32], ALD), s0);
      s1 = wmma_bf16(qf[kk], frag_b16n(&Ks[cur][16 * ALD + kk * 32], ALD), s1);
    }

    float p0[8], p1[8], corr[8];
#pragma unroll
    for (int r = 0; r < 8; ++r) {
      const int qrow = qclamp + r + 8 * half;
      const unsigned char* arl = &AdjS[cur][(qrow - qblk) * 32];
      const float b0 = (mt + nn      < m) ? alpha * (1.0f - (float)arl[nn])      : -1e30f;
      const float b1 = (mt + nn + 16 < m) ? alpha * (1.0f - (float)arl[nn + 16]) : -1e30f;
      float v0 = s0[r] * scale + b0;
      float v1 = s1[r] * scale + b1;
      float mx = fmaxf(v0, v1);
#pragma unroll
      for (int off = 1; off < 16; off <<= 1) mx = fmaxf(mx, __shfl_xor(mx, off, 32));
      const float mn = fmaxf(mst[r], mx);
      corr[r] = __expf(mst[r] - mn);
      p0[r] = __expf(v0 - mn);
      p1[r] = __expf(v1 - mn);
      mst[r] = mn;
    }

#pragma unroll
    for (int j = 0; j < 8; ++j)
#pragma unroll
      for (int r = 0; r < 8; ++r) oacc[j][r] *= corr[r];
#pragma unroll
    for (int r = 0; r < 8; ++r) lacc[r] *= corr[r];

#pragma unroll
    for (int r = 0; r < 8; ++r) {
      union { bf16_t hh[2]; unsigned int u; } pk;
      pk.hh[0] = (bf16_t)p0[r];
      pk.hh[1] = (bf16_t)p1[r];
      myP32[(r + 8 * half) * 16 + nn] = pk.u;
    }
    const v16bf pa = frag_a16x32(myP, 32);
    lacc = wmma_bf16(pa, onesf, lacc);           // row sums via matrix unit
#pragma unroll
    for (int j = 0; j < 8; ++j)
      oacc[j] = wmma_bf16(pa, frag_b16n(&VsT[cur][(j * 16) * VLD], VLD), oacc[j]);

    __syncthreads();
  }

  if (qvalid) {
#pragma unroll
    for (int j = 0; j < 8; ++j) {
#pragma unroll
      for (int r = 0; r < 8; ++r) {
        const int row = qbase + r + 8 * half;
        const int c   = j * 16 + nn;
        const float q = (float)Qhead[(size_t)row * 512 + c];
        const float v = q + oacc[j][r] / lacc[r];
        O[((size_t)b * n + row) * 512 + h * 128 + c] = (bf16_t)v;
      }
    }
  }
}

// ---------------------------------------------------------------- host driver
extern "C" void kernel_launch(void* const* d_in, const int* in_sizes, int n_in,
                              void* d_out, int out_size, void* d_ws, size_t ws_size,
                              hipStream_t stream) {
  (void)in_sizes; (void)n_in; (void)out_size; (void)ws_size;
  const int B = 4, V = 2048, NADJ = 2064, DIN = 192, DH = 512, SF = 196;
  const int MV = B * V, MS = B * 16;
  const int KP_SF = 224;

  const float* x    = (const float*)d_in[0];
  const float* adjf = (const float*)d_in[1];
  const float* seed = (const float*)d_in[2];
  const float* P[44];
  for (int i = 0; i < 44; ++i) P[i] = (const float*)d_in[3 + i];

  char* wsp = (char*)d_ws;
  auto alloc = [&](size_t bytes) -> void* {
    void* p = (void*)wsp;
    wsp += (bytes + 255) & ~(size_t)255;
    return p;
  };
  auto cvt = [&](const float* src, size_t nelem) -> bf16_t* {
    bf16_t* dst = (bf16_t*)alloc(nelem * sizeof(bf16_t));
    k_cvt<<<dim3((unsigned)((nelem + 255) / 256)), 256, 0, stream>>>(src, dst, (int)nelem);
    return dst;
  };
  auto cvtT = [&](const float* src, int K, int N, int Kpad) -> bf16_t* {
    const size_t nelem = (size_t)N * Kpad;
    bf16_t* dst = (bf16_t*)alloc(nelem * sizeof(bf16_t));
    k_cvtT<<<dim3((unsigned)((nelem + 255) / 256)), 256, 0, stream>>>(src, dst, K, N, Kpad);
    return dst;
  };

  bf16_t* Xbf = cvt(x, (size_t)MV * DIN);
  bf16_t* Sd = (bf16_t*)alloc((size_t)MS * KP_SF * 2);
  k_cvtPad<<<dim3((MS * KP_SF + 255) / 256), 256, 0, stream>>>(seed, Sd, MS, SF, KP_SF);
  const size_t adjN2 = (size_t)B * NADJ * NADJ;
  unsigned char* adj8 = (unsigned char*)alloc(adjN2);
  k_adj8<<<dim3((unsigned)((adjN2 + 255) / 256)), 256, 0, stream>>>(adjf, adj8, (int)adjN2);

  bf16_t* We0q = cvtT(P[0],  DIN, DH, DIN);
  bf16_t* We0k = cvtT(P[2],  DIN, DH, DIN);
  bf16_t* We0v = cvtT(P[4],  DIN, DH, DIN);
  bf16_t* We0o = cvtT(P[6],  DH, DH, DH);
  bf16_t* We1q = cvtT(P[8],  DH, DH, DH);
  bf16_t* We1k = cvtT(P[10], DH, DH, DH);
  bf16_t* We1v = cvtT(P[12], DH, DH, DH);
  bf16_t* We1o = cvtT(P[14], DH, DH, DH);
  bf16_t* WiS  = cvtT(P[16], SF, DH, KP_SF);
  bf16_t* Wpq  = cvtT(P[18], DH, DH, DH);
  bf16_t* Wpk  = cvtT(P[20], DH, DH, DH);
  bf16_t* Wpv  = cvtT(P[22], DH, DH, DH);
  bf16_t* Wpo  = cvtT(P[24], DH, DH, DH);
  bf16_t* W1q  = cvtT(P[26], DH, DH, DH);
  bf16_t* W1k  = cvtT(P[28], DH, DH, DH);
  bf16_t* W1v  = cvtT(P[30], DH, DH, DH);
  bf16_t* W1o  = cvtT(P[32], DH, DH, DH);
  bf16_t* W2q  = cvtT(P[34], DH, DH, DH);
  bf16_t* W2k  = cvtT(P[36], DH, DH, DH);
  bf16_t* W2v  = cvtT(P[38], DH, DH, DH);
  bf16_t* W2o  = cvtT(P[40], DH, DH, DH);
  bf16_t* Wout = cvtT(P[42], DH, DIN, DH);

  bf16_t* Qb   = (bf16_t*)alloc((size_t)MV * DH * 2);
  bf16_t* Kb   = (bf16_t*)alloc((size_t)MV * DH * 2);
  bf16_t* VT   = (bf16_t*)alloc((size_t)B * DH * 2048 * 2);
  bf16_t* Ob   = (bf16_t*)alloc((size_t)MV * DH * 2);
  bf16_t* H1   = (bf16_t*)alloc((size_t)MV * DH * 2);
  bf16_t* H2   = (bf16_t*)alloc((size_t)MV * DH * 2);
  bf16_t* Sb   = (bf16_t*)alloc((size_t)MS * DH * 2);
  bf16_t* Qs   = (bf16_t*)alloc((size_t)MS * DH * 2);
  bf16_t* Kd   = (bf16_t*)alloc((size_t)MS * DH * 2);
  bf16_t* VTd  = (bf16_t*)alloc((size_t)B * DH * 32 * 2);
  bf16_t* Os   = (bf16_t*)alloc((size_t)MS * DH * 2);
  bf16_t* D0   = (bf16_t*)alloc((size_t)MS * DH * 2);
  bf16_t* D1   = (bf16_t*)alloc((size_t)MS * DH * 2);
  bf16_t* D2   = (bf16_t*)alloc((size_t)MS * DH * 2);

  k_zero16<<<dim3((B * DH * 32 + 255) / 256), 256, 0, stream>>>(VTd, B * DH * 32);

  auto gemm_bf = [&](const bf16_t* A, const bf16_t* Wt, const float* bias,
                     bf16_t* Cbf, int M, int N, int K) {
    k_gemm<0><<<dim3((M + 127) / 128, N / 64), 128, 0, stream>>>(
        A, Wt, bias, nullptr, Cbf, nullptr, nullptr, 0, 0, M, N, K);
  };
  auto gemm_rr = [&](const bf16_t* A, const bf16_t* Wt, const float* bias,
                     const bf16_t* resid, bf16_t* Cbf, int M, int N, int K) {
    k_gemm<1><<<dim3((M + 127) / 128, N / 64), 128, 0, stream>>>(
        A, Wt, bias, resid, Cbf, nullptr, nullptr, 0, 0, M, N, K);
  };
  auto gemm_f32 = [&](const bf16_t* A, const bf16_t* Wt, const float* bias,
                      float* Cf, int M, int N, int K) {
    k_gemm<2><<<dim3((M + 127) / 128, N / 64), 128, 0, stream>>>(
        A, Wt, bias, nullptr, nullptr, Cf, nullptr, 0, 0, M, N, K);
  };
  auto gemm_vt = [&](const bf16_t* A, const bf16_t* Wt, const float* bias,
                     bf16_t* CvT, int kb_shift, int mpad, int M, int N, int K) {
    k_gemm<3><<<dim3((M + 127) / 128, N / 64), 128, 0, stream>>>(
        A, Wt, bias, nullptr, nullptr, nullptr, CvT, kb_shift, mpad, M, N, K);
  };

  // ---- enc0
  gemm_bf(Xbf, We0q, P[1], Qb, MV, DH, DIN);
  gemm_bf(Xbf, We0k, P[3], Kb, MV, DH, DIN);
  gemm_vt(Xbf, We0v, P[5], VT, 11, 2048, MV, DH, DIN);
  k_attn<<<dim3(V / 64, 16), 128, 0, stream>>>(Qb, Kb, VT, adj8, Ob, V, V, 2048, NADJ, 0, 0);
  gemm_rr(Ob, We0o, P[7], Ob, H1, MV, DH, DH);

  // ---- enc1
  gemm_bf(H1, We1q, P[9],  Qb, MV, DH, DH);
  gemm_bf(H1, We1k, P[11], Kb, MV, DH, DH);
  gemm_vt(H1, We1v, P[13], VT, 11, 2048, MV, DH, DH);
  k_attn<<<dim3(V / 64, 16), 128, 0, stream>>>(Qb, Kb, VT, adj8, Ob, V, V, 2048, NADJ, 0, 0);
  gemm_rr(Ob, We1o, P[15], Ob, H2, MV, DH, DH);

  // ---- S = lin(inputS, seed)
  gemm_bf(Sd, WiS, P[17], Sb, MS, DH, KP_SF);

  // ---- pma
  gemm_bf(Sb, Wpq, P[19], Qs, MS, DH, DH);
  gemm_bf(H2, Wpk, P[21], Kb, MV, DH, DH);
  gemm_vt(H2, Wpv, P[23], VT, 11, 2048, MV, DH, DH);
  k_attn<<<dim3(1, 16), 128, 0, stream>>>(Qs, Kb, VT, adj8, Os, 16, V, 2048, NADJ, V, 0);
  gemm_rr(Os, Wpo, P[25], Os, D0, MS, DH, DH);

  // ---- dec1
  gemm_bf(D0, W1q, P[27], Qs, MS, DH, DH);
  gemm_bf(D0, W1k, P[29], Kd, MS, DH, DH);
  gemm_vt(D0, W1v, P[31], VTd, 4, 32, MS, DH, DH);
  k_attn<<<dim3(1, 16), 128, 0, stream>>>(Qs, Kd, VTd, adj8, Os, 16, 16, 32, NADJ, V, V);
  gemm_rr(Os, W1o, P[33], Os, D1, MS, DH, DH);

  // ---- dec2
  gemm_bf(D1, W2q, P[35], Qs, MS, DH, DH);
  gemm_bf(D1, W2k, P[37], Kd, MS, DH, DH);
  gemm_vt(D1, W2v, P[39], VTd, 4, 32, MS, DH, DH);
  k_attn<<<dim3(1, 16), 128, 0, stream>>>(Qs, Kd, VTd, adj8, Os, 16, 16, 32, NADJ, V, V);
  gemm_rr(Os, W2o, P[41], Os, D2, MS, DH, DH);

  // ---- out
  gemm_f32(D2, Wout, P[43], (float*)d_out, MS, DIN, DH);
}
